// TwoLayerRNN_58634893525214
// MI455X (gfx1250) — matrix operand order repaired and checked by reading, not compiled, not run on hardware
//
#include <hip/hip_runtime.h>
#include <hip/hip_bf16.h>

// Problem constants (match reference)
#define TT  512
#define BB  128
#define II  256
#define HH  1024
#define NWG 128          // 64 WGs for layer1 tiles, 64 WGs for layer2 tiles
#define A1  0.01f
#define A2  0.01f

typedef __bf16 bf16_t;
typedef __attribute__((ext_vector_type(16))) __bf16 v16bf;
typedef __attribute__((ext_vector_type(8)))  __bf16 v8bf;
typedef __attribute__((ext_vector_type(8)))  float  v8f;
typedef __attribute__((ext_vector_type(4)))  float  v4f;

union V16 { v16bf v; v8bf h[2]; };

__device__ __forceinline__ v8f wmma_bf16(v16bf a, v16bf b, v8f c) {
  // (neg_a, A, neg_b, B, c_mod, C, reuse_a, reuse_b)
  return __builtin_amdgcn_wmma_f32_16x16x32_bf16(false, a, false, b, (short)0, c,
                                                 false, false);
}

// A operand (16x32 bf16, MxK): lane = (m, kh); elems 0-7 -> K = k0+kh*8..,
// elems 8-15 -> K = k0+16+kh*8.. (per CDNA5 ISA 16-bit A layout)
__device__ __forceinline__ v16bf loadA_bf(const bf16_t* rowptr, int k0, int kh) {
  const bf16_t* p = rowptr + k0 + kh * 8;
  V16 u;
  u.h[0] = *(const v8bf*)(p);
  u.h[1] = *(const v8bf*)(p + 16);
  return u.v;
}

// A operand built from fp32 source (x) with on-the-fly bf16 convert
__device__ __forceinline__ v16bf loadA_f32(const float* rowptr, int k0, int kh) {
  const float* p = rowptr + k0 + kh * 8;
  v4f a0 = *(const v4f*)(p);
  v4f a1 = *(const v4f*)(p + 4);
  v4f b0 = *(const v4f*)(p + 16);
  v4f b1 = *(const v4f*)(p + 20);
  V16 u;
  #pragma unroll
  for (int i = 0; i < 4; ++i) {
    u.h[0][i]     = (bf16_t)a0[i];
    u.h[0][i + 4] = (bf16_t)a1[i];
    u.h[1][i]     = (bf16_t)b0[i];
    u.h[1][i + 4] = (bf16_t)b1[i];
  }
  return u.v;
}

// B operand (32x16 bf16, KxN): B[k][n] = W[n][k]; lane = (n, kh); elems 0-7 -> K = k0+kh*8..,
// elems 8-15 -> K = k0+16+kh*8.. (same K order as the A operand).
// Reads from LDS -> ds_load_b128.
__device__ __forceinline__ v16bf loadB(const bf16_t* base, int ld, int n, int k0, int kh) {
  const bf16_t* p = base + n * ld + k0 + kh * 8;
  V16 u;
  u.h[0] = *(const v8bf*)(p);
  u.h[1] = *(const v8bf*)(p + 16);
  return u.v;
}

__device__ __forceinline__ void grid_barrier(unsigned* cnt, unsigned phase) {
  __syncthreads();
  if (threadIdx.x == 0) {
    __hip_atomic_fetch_add(cnt, 1u, __ATOMIC_RELEASE, __HIP_MEMORY_SCOPE_AGENT);
    const unsigned target = phase * (unsigned)NWG;
    while (__hip_atomic_load(cnt, __ATOMIC_ACQUIRE, __HIP_MEMORY_SCOPE_AGENT) < target)
      __builtin_amdgcn_s_sleep(1);
  }
  __syncthreads();
  __threadfence();   // agent-scope acquire: invalidate stale near-cache lines
}

__global__ void rnn_prep(const float* __restrict__ h1_0, const float* __restrict__ h2_0,
                         bf16_t* __restrict__ h1bf, bf16_t* __restrict__ h2bf, int n) {
  int i = blockIdx.x * blockDim.x + threadIdx.x;
  if (i < n) {
    h1bf[i] = (bf16_t)h1_0[i];   // buffer 0 = state entering step 0
    h2bf[i] = (bf16_t)h2_0[i];
  }
}

__global__ __launch_bounds__(256)
void rnn_persistent(const float* __restrict__ x,
                    const float* __restrict__ h1_0, const float* __restrict__ h2_0,
                    const float* __restrict__ Wi1, const float* __restrict__ bi1,
                    const float* __restrict__ Wh1, const float* __restrict__ bh1,
                    const float* __restrict__ Wi2, const float* __restrict__ bi2,
                    const float* __restrict__ Wh2, const float* __restrict__ bh2,
                    float* __restrict__ out,
                    unsigned* __restrict__ bar,
                    bf16_t* __restrict__ h1bf, bf16_t* __restrict__ h2bf) {
  // 64 KB LDS: per-WG weight slice (16 output rows), bf16, resident all 512 steps
  __shared__ bf16_t smA[16 * HH];   // L1: Wi1 rows (16xII)   L2: Wi2 rows (16xHH)
  __shared__ bf16_t smB[16 * HH];   // L1: Wh1 rows (16xHH)   L2: Wh2 rows (16xHH)

  const int g     = blockIdx.x;
  const bool isL2 = (g >= 64);
  const int nb    = isL2 ? (g - 64) : g;         // output-column tile 0..63

  // ---- stage + convert this WG's weight rows into LDS (rows are contiguous) ----
  {
    const float* WA = isL2 ? Wi2 : Wi1;
    const float* WB = isL2 ? Wh2 : Wh1;
    const int KA = isL2 ? HH : II;
    const size_t offA = (size_t)nb * 16 * KA;
    const size_t offB = (size_t)nb * 16 * HH;
    for (int i = threadIdx.x; i < 16 * KA; i += 256) smA[i] = (bf16_t)WA[offA + i];
    for (int i = threadIdx.x; i < 16 * HH; i += 256) smB[i] = (bf16_t)WB[offB + i];
  }
  __syncthreads();

  const int lane = threadIdx.x & 31;
  const int wv   = threadIdx.x >> 5;    // 0..7 : batch-row tile (mb)
  const int m    = lane & 15;           // A row within tile / B,C column within tile
  const int kh   = lane >> 4;           // K-half selector
  const int col  = nb * 16 + m;         // global output column

  const float bias = isL2 ? (bi2[col] + bh2[col]) : (bi1[col] + bh1[col]);

  float* __restrict__ out1 = out;
  float* __restrict__ out2 = out + (size_t)TT * BB * HH;
  float* __restrict__ h1f  = out + 2 * (size_t)TT * BB * HH;
  float* __restrict__ h2f  = h1f + (size_t)BB * HH;

  unsigned phase = 0;
  for (int t = 0; t <= TT; ++t) {
    if (!isL2) {
      if (t < TT) {
        // ---------------- layer 1, step t ----------------
        v8f acc = {};
        const float* xrow = x + ((size_t)t * BB + (wv * 16 + m)) * II;
        #pragma unroll
        for (int k0 = 0; k0 < II; k0 += 32)
          acc = wmma_bf16(loadA_f32(xrow, k0, kh), loadB(smA, II, m, k0, kh), acc);

        const bf16_t* h1r = h1bf + (size_t)(t & 1) * BB * HH + (size_t)(wv * 16 + m) * HH;
        #pragma unroll 4
        for (int k0 = 0; k0 < HH; k0 += 32)
          acc = wmma_bf16(loadA_bf(h1r, k0, kh), loadB(smB, HH, m, k0, kh), acc);

        const float* prev = (t == 0) ? h1_0 : (out1 + (size_t)(t - 1) * BB * HH);
        float*  o  = out1 + (size_t)t * BB * HH;
        bf16_t* hw = h1bf + (size_t)((t + 1) & 1) * BB * HH;
        #pragma unroll
        for (int r = 0; r < 8; ++r) {
          const size_t idx = (size_t)(wv * 16 + kh * 8 + r) * HH + col;
          float hn = acc[r] + bias;
          hn = hn > 0.f ? hn : 0.f;
          const float nv = (1.0f - A1) * prev[idx] + A1 * hn;
          o[idx]  = nv;
          hw[idx] = (bf16_t)nv;
        }
      }
    } else {
      const int u = t - 1;                 // layer 2 runs one step behind (pipelined)
      if (u >= 0) {
        // ---------------- layer 2, step u ----------------
        v8f acc = {};
        // h1n(u) == state entering step t == h1bf[t&1]
        const bf16_t* h1r = h1bf + (size_t)(t & 1) * BB * HH + (size_t)(wv * 16 + m) * HH;
        #pragma unroll 4
        for (int k0 = 0; k0 < HH; k0 += 32)
          acc = wmma_bf16(loadA_bf(h1r, k0, kh), loadB(smA, HH, m, k0, kh), acc);

        const bf16_t* h2r = h2bf + (size_t)(u & 1) * BB * HH + (size_t)(wv * 16 + m) * HH;
        #pragma unroll 4
        for (int k0 = 0; k0 < HH; k0 += 32)
          acc = wmma_bf16(loadA_bf(h2r, k0, kh), loadB(smB, HH, m, k0, kh), acc);

        const float* prev = (u == 0) ? h2_0 : (out2 + (size_t)(u - 1) * BB * HH);
        float*  o  = out2 + (size_t)u * BB * HH;
        bf16_t* hw = h2bf + (size_t)((u + 1) & 1) * BB * HH;
        #pragma unroll
        for (int r = 0; r < 8; ++r) {
          const size_t idx = (size_t)(wv * 16 + kh * 8 + r) * HH + col;
          float hn = acc[r] + bias;
          hn = hn > 0.f ? hn : 0.f;
          const float nv = (1.0f - A2) * prev[idx] + A2 * hn;
          o[idx]  = nv;
          hw[idx] = (bf16_t)nv;
        }
      }
    }
    ++phase;
    grid_barrier(bar, phase);
  }

  // final hidden states = last-step outputs
  const size_t n = (size_t)BB * HH;
  for (size_t i = blockIdx.x * 256 + threadIdx.x; i < n; i += (size_t)NWG * 256) {
    h1f[i] = out1[(size_t)(TT - 1) * BB * HH + i];
    h2f[i] = out2[(size_t)(TT - 1) * BB * HH + i];
  }
}

extern "C" void kernel_launch(void* const* d_in, const int* in_sizes, int n_in,
                              void* d_out, int out_size, void* d_ws, size_t ws_size,
                              hipStream_t stream) {
  const float* x    = (const float*)d_in[0];
  const float* h1_0 = (const float*)d_in[1];
  const float* h2_0 = (const float*)d_in[2];
  const float* Wi1  = (const float*)d_in[3];
  const float* bi1  = (const float*)d_in[4];
  const float* Wh1  = (const float*)d_in[5];
  const float* bh1  = (const float*)d_in[6];
  const float* Wi2  = (const float*)d_in[7];
  const float* bi2  = (const float*)d_in[8];
  const float* Wh2  = (const float*)d_in[9];
  const float* bh2  = (const float*)d_in[10];

  char* ws = (char*)d_ws;
  unsigned* bar = (unsigned*)ws;                                  // 256 B
  bf16_t* h1bf = (bf16_t*)(ws + 256);                             // 2 x B x H bf16
  bf16_t* h2bf = (bf16_t*)(ws + 256 + 2 * (size_t)BB * HH * 2);   // 2 x B x H bf16

  hipMemsetAsync(ws, 0, 256, stream);                 // reset barrier each replay
  const int n = BB * HH;
  rnn_prep<<<(n + 255) / 256, 256, 0, stream>>>(h1_0, h2_0, h1bf, h2bf, n);
  rnn_persistent<<<NWG, 256, 0, stream>>>(x, h1_0, h2_0,
                                          Wi1, bi1, Wh1, bh1,
                                          Wi2, bi2, Wh2, bh2,
                                          (float*)d_out, bar, h1bf, h2bf);
}
